// SelfAttention_66520453480452
// MI455X (gfx1250) — matrix operand order repaired and checked by reading, not compiled, not run on hardware
//
#include <hip/hip_runtime.h>

// MI455X / gfx1250, wave32, WMMA f16->f32.
// Pipeline: [w cvt+swizzle f32->f16] -> [qkv+rmsnorm+rope] -> [flash attn]
//           -> [out proj].

#define B_   2
#define S_   2048
#define DIM_ 1536
#define NH_  12
#define HD_  128

typedef __attribute__((ext_vector_type(16))) _Float16 v16h;
typedef __attribute__((ext_vector_type(8)))  _Float16 v8h;
typedef __attribute__((ext_vector_type(8)))  float    v8f;

__device__ __forceinline__ v8f wmma_f16(v16h a, v16h b, v8f c) {
  // D = A(16x32 f16) * B(32x16 f16) + C(16x16 f32)
  return __builtin_amdgcn_wmma_f32_16x16x32_f16(false, a, false, b, (short)0, c,
                                                false, false);
}

__device__ __forceinline__ v16h join16(v8h lo, v8h hi) {
  return __builtin_shufflevector(lo, hi, 0, 1, 2, 3, 4, 5, 6, 7,
                                 8, 9, 10, 11, 12, 13, 14, 15);
}

// A fragment (16x32, f16) from LDS row-major [rows][ld] (ld*2 % 16 == 0).
// lane m = L%16, element e -> k = e + 16*(L/16)
__device__ __forceinline__ v16h lds_load_A(const _Float16* base, int row0, int k0, int ld) {
  const int lane = threadIdx.x & 31;
  const int m = lane & 15, g = lane >> 4;
  const _Float16* p = base + (row0 + m) * ld + k0 + 16 * g;
  return join16(*(const v8h*)p, *(const v8h*)(p + 8));
}

// B fragment (32x16, f16) from LDS: element (k,n) at base[(n0+n)*ld + k0 + k],
// lane n = L%16, element e -> k = e + 16*(L/16). Two aligned 16B loads.
__device__ __forceinline__ v16h lds_load_B(const _Float16* base, int n0, int k0, int ld) {
  const int lane = threadIdx.x & 31;
  const int n = lane & 15, g = lane >> 4;
  const _Float16* p = base + (n0 + n) * ld + k0 + 16 * g;
  return join16(*(const v8h*)p, *(const v8h*)(p + 8));
}

// B fragment from swizzled f16 weights Wsw[k/32][n][32] (fragment-major):
// each lane's 32B slice is contiguous; 32 lanes cover one contiguous 1KB block.
__device__ __forceinline__ v16h glb_load_Bsw(const _Float16* __restrict__ w,
                                             int n0, int k0) {
  const int lane = threadIdx.x & 31;
  const int n = lane & 15, g = lane >> 4;
  const _Float16* p =
      w + (((size_t)(k0 >> 5) * DIM_ + (n0 + n)) << 5) + 16 * g;
  return join16(*(const v8h*)p, *(const v8h*)(p + 8));
}

// ---------------------------------------------------------------------------
// Kernel 0: one-time weight f32 -> f16 conversion + fragment-major swizzle.
// src: [DIM][DIM] f32 row-major (n,k). dst: [DIM/32][DIM][32] f16.
// ---------------------------------------------------------------------------
__global__ __launch_bounds__(256) void cvt_w_swizzle_kernel(
    const float* __restrict__ src, _Float16* __restrict__ dst) {
  int base = (blockIdx.x * 256 + threadIdx.x) * 4;  // 4 consecutive k
  int n = base / DIM_;
  int k = base - n * DIM_;
  float4 v = *(const float4*)(src + base);
  _Float16* d = dst + (((size_t)(k >> 5) * DIM_ + n) << 5) + (k & 31);
  d[0] = (_Float16)v.x;
  d[1] = (_Float16)v.y;
  d[2] = (_Float16)v.z;
  d[3] = (_Float16)v.w;
}

// ---------------------------------------------------------------------------
// Shared GEMM core: 32 tokens (LDS, f16) x W^T (swizzled f16, global/L2),
// each wave computes 2 row-tiles x 4 col-tiles per pass; result -> LDS f32.
// Called with a direct kernel-arg pointer so W loads stay global_load.
// ---------------------------------------------------------------------------
#define XLD 1544  // padded half stride for 1536-wide LDS tiles (16B aligned)

__device__ __forceinline__ void gemm32(const _Float16* xt,
                                       const _Float16* __restrict__ W,
                                       float* res, int wave, int col, int g) {
  for (int grp = wave; grp < 24; grp += 8) {
    const int ct0 = grp * 4;
    v8f acc[2][4] = {};
    for (int k0 = 0; k0 < DIM_; k0 += 32) {
      v16h a0 = lds_load_A(xt, 0, k0, XLD);
      v16h a1 = lds_load_A(xt, 16, k0, XLD);
      v16h bf0 = glb_load_Bsw(W, (ct0 + 0) * 16, k0);
      v16h bf1 = glb_load_Bsw(W, (ct0 + 1) * 16, k0);
      v16h bf2 = glb_load_Bsw(W, (ct0 + 2) * 16, k0);
      v16h bf3 = glb_load_Bsw(W, (ct0 + 3) * 16, k0);
      acc[0][0] = wmma_f16(a0, bf0, acc[0][0]);
      acc[1][0] = wmma_f16(a1, bf0, acc[1][0]);
      acc[0][1] = wmma_f16(a0, bf1, acc[0][1]);
      acc[1][1] = wmma_f16(a1, bf1, acc[1][1]);
      acc[0][2] = wmma_f16(a0, bf2, acc[0][2]);
      acc[1][2] = wmma_f16(a1, bf2, acc[1][2]);
      acc[0][3] = wmma_f16(a0, bf3, acc[0][3]);
      acc[1][3] = wmma_f16(a1, bf3, acc[1][3]);
    }
#pragma unroll
    for (int rt = 0; rt < 2; ++rt)
#pragma unroll
      for (int t = 0; t < 4; ++t)
#pragma unroll
        for (int vr = 0; vr < 8; ++vr)
          res[(rt * 16 + vr + 8 * g) * DIM_ + (ct0 + t) * 16 + col] =
              acc[rt][t][vr];
  }
}

// ---------------------------------------------------------------------------
// Kernel 1: q/k/v = proj(x); q,k -> rmsnorm -> grid rope; store f16 [B,NH,S,HD]
// ---------------------------------------------------------------------------
__global__ __launch_bounds__(256) void qkv_rmsnorm_rope_kernel(
    const float* __restrict__ x,
    const _Float16* __restrict__ wqh, const float* __restrict__ bq,
    const _Float16* __restrict__ wkh, const float* __restrict__ bk,
    const _Float16* __restrict__ wvh, const float* __restrict__ bv,
    const float* __restrict__ nqw, const float* __restrict__ nkw,
    const float* __restrict__ cf, const float* __restrict__ sf,
    const float* __restrict__ ch, const float* __restrict__ sh,
    const float* __restrict__ cwt, const float* __restrict__ swt,
    _Float16* __restrict__ qout, _Float16* __restrict__ kout,
    _Float16* __restrict__ vout) {
  extern __shared__ char smem[];
  _Float16* xt  = (_Float16*)smem;                       // [32][XLD] f16
  float*    res = (float*)(smem + 32 * XLD * 2);         // [32][1536] f32
  float*    red = res + 32 * DIM_;                       // [32][8]

  const int tid = threadIdx.x;
  const int token0 = blockIdx.x * 32;  // 32 tokens per WG, same batch
  const int b = token0 / S_;

  // stage x tile, f32 -> f16
  for (int i = tid; i < 32 * DIM_; i += 256) {
    int r = i / DIM_, c = i - r * DIM_;
    xt[r * XLD + c] = (_Float16)x[(size_t)(token0 + r) * DIM_ + c];
  }
  __syncthreads();

  const int wave = tid >> 5;
  const int lane = tid & 31;
  const int col = lane & 15, g = lane >> 4;

  for (int mi = 0; mi < 3; ++mi) {
    // direct-arg pointer at each call site keeps weight loads in global space
    if (mi == 0)      gemm32(xt, wqh, res, wave, col, g);
    else if (mi == 1) gemm32(xt, wkh, res, wave, col, g);
    else              gemm32(xt, wvh, res, wave, col, g);
    __syncthreads();

    const float* bias = (mi == 0) ? bq : (mi == 1) ? bk : bv;

    if (mi < 2) {
      // RMS over full 1536-dim row (f32): 8 partial lanes per row
      int rr = tid >> 3, sub = tid & 7;
      float p = 0.f;
      for (int c = sub; c < DIM_; c += 8) {
        float v = res[rr * DIM_ + c] + bias[c];
        p += v * v;
      }
      red[rr * 8 + sub] = p;
      __syncthreads();
      if (sub == 0) {
        float s = 0.f;
#pragma unroll
        for (int j = 0; j < 8; ++j) s += red[rr * 8 + j];
        red[rr * 8] = rsqrtf(s * (1.0f / DIM_) + 1e-6f);
      }
      __syncthreads();

      const float* nw = (mi == 0) ? nqw : nkw;
      _Float16* out = (mi == 0) ? qout : kout;
      // 32 rows * 12 heads * 64 rotary pairs
      for (int p2 = tid; p2 < 32 * NH_ * 64; p2 += 256) {
        int r = p2 / (NH_ * 64);
        int rem = p2 - r * (NH_ * 64);
        int head = rem >> 6;
        int c = rem & 63;
        int sidx = (token0 + r) & (S_ - 1);
        int fi = sidx >> 8;            // /(16*16)
        int hi = (sidx >> 4) & 15;
        int wi = sidx & 15;
        float cv, sv;
        if (c < 22)       { cv = cf[fi * 22 + c];         sv = sf[fi * 22 + c]; }
        else if (c < 43)  { cv = ch[hi * 21 + (c - 22)];  sv = sh[hi * 21 + (c - 22)]; }
        else              { cv = cwt[wi * 21 + (c - 43)]; sv = swt[wi * 21 + (c - 43)]; }
        int d0 = head * HD_ + 2 * c;
        float rms = red[r * 8];
        float xr = (res[r * DIM_ + d0]     + bias[d0])     * rms * nw[d0];
        float xi = (res[r * DIM_ + d0 + 1] + bias[d0 + 1]) * rms * nw[d0 + 1];
        float orr = xr * cv - xi * sv;
        float ori = xr * sv + xi * cv;
        size_t off = ((size_t)(b * NH_ + head) * S_ + sidx) * HD_ + 2 * c;
        out[off]     = (_Float16)orr;
        out[off + 1] = (_Float16)ori;
      }
    } else {
      // v: bias only, store [B,NH,S,HD] f16
      for (int i = tid; i < 32 * DIM_; i += 256) {
        int r = i / DIM_, c = i - r * DIM_;
        int head = c >> 7, hd = c & 127;
        int sidx = (token0 + r) & (S_ - 1);
        float v = res[r * DIM_ + c] + bias[c];
        vout[((size_t)(b * NH_ + head) * S_ + sidx) * HD_ + hd] = (_Float16)v;
      }
    }
    __syncthreads();
  }
}

// ---------------------------------------------------------------------------
// Kernel 2: flash attention, 64 queries per WG, 4 waves (16 q-rows each),
// key tiles of 32, online softmax, ragged key-length mask.
// ---------------------------------------------------------------------------
#define QLD 136
#define KLD 136
#define VLD 40
#define PLD 40

__global__ __launch_bounds__(128) void flash_attn_kernel(
    const _Float16* __restrict__ q, const _Float16* __restrict__ k,
    const _Float16* __restrict__ v, const int* __restrict__ seq_lens,
    _Float16* __restrict__ aout) {
  __shared__ _Float16 Qs[64 * QLD];
  __shared__ _Float16 Ks[32 * KLD];
  __shared__ _Float16 Vt[128 * VLD];      // V transposed: [hd][key]
  __shared__ _Float16 Ps[4][16 * PLD];    // per-wave P scratch (C->A relayout)

  const int tid = threadIdx.x;
  const int wave = tid >> 5;
  const int lane = tid & 31;
  const int col = lane & 15, g = lane >> 4;

  const int bh = blockIdx.x;                 // b*NH + head
  const int b = bh / NH_;
  const int head = bh - b * NH_;
  const int q0 = blockIdx.y * 64;
  const int slen = seq_lens[b];

  const _Float16* qg  = q + ((size_t)bh * S_ + q0) * HD_;
  const _Float16* kg0 = k + (size_t)bh * S_ * HD_;
  const _Float16* vg0 = v + (size_t)bh * S_ * HD_;

  // stage Q (64x128) via 16B chunks
  for (int c = tid; c < 1024; c += 128) {
    int r = c >> 4, hdc = (c & 15) << 3;
    *(uint4*)(&Qs[r * QLD + hdc]) = *(const uint4*)(&qg[r * HD_ + hdc]);
  }
  __syncthreads();

  const int r0 = wave * 16;
  v16h aq[4];
#pragma unroll
  for (int k4 = 0; k4 < 4; ++k4) aq[k4] = lds_load_A(Qs, r0, k4 * 32, QLD);

  v8f oacc[8] = {};
  float mrow[8], lrow[8];
#pragma unroll
  for (int vr = 0; vr < 8; ++vr) { mrow[vr] = -3.0e38f; lrow[vr] = 0.f; }

  const float sscale = 0.08838834764831845f;  // 1/sqrt(128)
  const int nkt = (slen + 31) >> 5;           // fully-masked tiles contribute 0

  for (int kt = 0; kt < nkt; ++kt) {
    // stage K tile (32x128) row-major
    for (int c = tid; c < 512; c += 128) {
      int r = c >> 4, hdc = (c & 15) << 3;
      *(uint4*)(&Ks[r * KLD + hdc]) =
          *(const uint4*)(&kg0[(size_t)(kt * 32 + r) * HD_ + hdc]);
    }
    // stage V tile transposed [hd][key]
    for (int i = tid; i < 4096; i += 128) {
      int key = i >> 7, hd = i & 127;
      Vt[hd * VLD + key] = vg0[(size_t)(kt * 32 + key) * HD_ + hd];
    }
    __syncthreads();

    // S = Q * K^T : two 16x16 tiles per wave, HD reduction in 4 wmma steps
    v8f s0 = {}, s1 = {};
#pragma unroll
    for (int k4 = 0; k4 < 4; ++k4) {
      s0 = wmma_f16(aq[k4], lds_load_B(Ks, 0,  k4 * 32, KLD), s0);
      s1 = wmma_f16(aq[k4], lds_load_B(Ks, 16, k4 * 32, KLD), s1);
    }
    const bool ok0 = (kt * 32 + col) < slen;
    const bool ok1 = (kt * 32 + 16 + col) < slen;

    _Float16* Pw = Ps[wave];
#pragma unroll
    for (int vr = 0; vr < 8; ++vr) {
      float a0 = ok0 ? s0[vr] * sscale : -1.0e30f;
      float a1 = ok1 ? s1[vr] * sscale : -1.0e30f;
      // row max/sum across the 16 lanes that hold this row's columns
      float rmax = fmaxf(a0, a1);
      rmax = fmaxf(rmax, __shfl_xor(rmax, 1, 16));
      rmax = fmaxf(rmax, __shfl_xor(rmax, 2, 16));
      rmax = fmaxf(rmax, __shfl_xor(rmax, 4, 16));
      rmax = fmaxf(rmax, __shfl_xor(rmax, 8, 16));
      float mold = mrow[vr];
      float mnew = fmaxf(mold, rmax);
      float p0 = __expf(a0 - mnew);
      float p1 = __expf(a1 - mnew);
      float rsum = p0 + p1;
      rsum += __shfl_xor(rsum, 1, 16);
      rsum += __shfl_xor(rsum, 2, 16);
      rsum += __shfl_xor(rsum, 4, 16);
      rsum += __shfl_xor(rsum, 8, 16);
      float corr = __expf(mold - mnew);
      lrow[vr] = lrow[vr] * corr + rsum;
      mrow[vr] = mnew;
#pragma unroll
      for (int ct = 0; ct < 8; ++ct) oacc[ct][vr] *= corr;
      Pw[(vr + 8 * g) * PLD + col]      = (_Float16)p0;
      Pw[(vr + 8 * g) * PLD + 16 + col] = (_Float16)p1;
    }

    // O += P(16x32) * V(32x128): one wmma per 16-wide hd strip, K=32 exact
    v16h ap = lds_load_A(Pw, 0, 0, PLD);
#pragma unroll
    for (int ct = 0; ct < 8; ++ct)
      oacc[ct] = wmma_f16(ap, lds_load_B(Vt, ct * 16, 0, VLD), oacc[ct]);

    __syncthreads();
  }

  // normalize and store token-major f16 [B,S,DIM] for the final projection
#pragma unroll
  for (int vr = 0; vr < 8; ++vr) {
    float inv = 1.0f / lrow[vr];
    int row = q0 + r0 + vr + 8 * g;
    size_t base = ((size_t)b * S_ + row) * DIM_ + head * HD_;
#pragma unroll
    for (int ct = 0; ct < 8; ++ct)
      aout[base + ct * 16 + col] = (_Float16)(oacc[ct][vr] * inv);
  }
}

// ---------------------------------------------------------------------------
// Kernel 3: out = attn @ wo^T + bo (f32 output), same 2x4 tile blocking
// ---------------------------------------------------------------------------
__global__ __launch_bounds__(256) void out_proj_kernel(
    const _Float16* __restrict__ a, const _Float16* __restrict__ woh,
    const float* __restrict__ bo, float* __restrict__ out) {
  extern __shared__ char smem[];
  _Float16* at = (_Float16*)smem;  // [32][XLD]
  const int tid = threadIdx.x;
  const int token0 = blockIdx.x * 32;

  for (int c = tid; c < 32 * (DIM_ / 8); c += 256) {
    int r = c / (DIM_ / 8);
    int cc = (c - r * (DIM_ / 8)) * 8;
    *(uint4*)(&at[r * XLD + cc]) =
        *(const uint4*)(&a[(size_t)(token0 + r) * DIM_ + cc]);
  }
  __syncthreads();

  const int wave = tid >> 5, lane = tid & 31;
  const int col = lane & 15, g = lane >> 4;

  for (int grp = wave; grp < 24; grp += 8) {
    const int ct0 = grp * 4;
    v8f acc[2][4] = {};
    for (int k0 = 0; k0 < DIM_; k0 += 32) {
      v16h a0 = lds_load_A(at, 0, k0, XLD);
      v16h a1 = lds_load_A(at, 16, k0, XLD);
      v16h bf0 = glb_load_Bsw(woh, (ct0 + 0) * 16, k0);
      v16h bf1 = glb_load_Bsw(woh, (ct0 + 1) * 16, k0);
      v16h bf2 = glb_load_Bsw(woh, (ct0 + 2) * 16, k0);
      v16h bf3 = glb_load_Bsw(woh, (ct0 + 3) * 16, k0);
      acc[0][0] = wmma_f16(a0, bf0, acc[0][0]);
      acc[1][0] = wmma_f16(a1, bf0, acc[1][0]);
      acc[0][1] = wmma_f16(a0, bf1, acc[0][1]);
      acc[1][1] = wmma_f16(a1, bf1, acc[1][1]);
      acc[0][2] = wmma_f16(a0, bf2, acc[0][2]);
      acc[1][2] = wmma_f16(a1, bf2, acc[1][2]);
      acc[0][3] = wmma_f16(a0, bf3, acc[0][3]);
      acc[1][3] = wmma_f16(a1, bf3, acc[1][3]);
    }
#pragma unroll
    for (int rt = 0; rt < 2; ++rt)
#pragma unroll
      for (int t = 0; t < 4; ++t) {
        float bb = bo[(ct0 + t) * 16 + col];
#pragma unroll
        for (int vr = 0; vr < 8; ++vr)
          out[(size_t)(token0 + rt * 16 + vr + 8 * g) * DIM_ +
              (ct0 + t) * 16 + col] = acc[rt][t][vr] + bb;
      }
  }
}

// ---------------------------------------------------------------------------
extern "C" void kernel_launch(void* const* d_in, const int* in_sizes, int n_in,
                              void* d_out, int out_size, void* d_ws, size_t ws_size,
                              hipStream_t stream) {
  (void)in_sizes; (void)n_in; (void)out_size; (void)ws_size;
  const float* x        = (const float*)d_in[0];
  const int*   seq_lens = (const int*)d_in[1];
  // d_in[2..4] = f,h,w (hardcoded 8/16/16; GF*GH*GW == S)
  const float* wq  = (const float*)d_in[5];
  const float* bq  = (const float*)d_in[6];
  const float* wk  = (const float*)d_in[7];
  const float* bk  = (const float*)d_in[8];
  const float* wv  = (const float*)d_in[9];
  const float* bv  = (const float*)d_in[10];
  const float* wo  = (const float*)d_in[11];
  const float* bo  = (const float*)d_in[12];
  const float* nqw = (const float*)d_in[13];
  const float* nkw = (const float*)d_in[14];
  const float* cf  = (const float*)d_in[15];
  const float* sf  = (const float*)d_in[16];
  const float* ch  = (const float*)d_in[17];
  const float* sh  = (const float*)d_in[18];
  const float* cw  = (const float*)d_in[19];
  const float* sw  = (const float*)d_in[20];

  const size_t N  = (size_t)B_ * NH_ * S_ * HD_;  // 6,291,456 halves per buffer
  const size_t WN = (size_t)DIM_ * DIM_;          // 2,359,296 halves per weight
  _Float16* qb  = (_Float16*)d_ws;
  _Float16* kb  = qb + N;
  _Float16* vb  = kb + N;
  _Float16* ab  = vb + N;   // attention output, token-major [B,S,DIM]
  _Float16* wqh = ab + N;
  _Float16* wkh = wqh + WN;
  _Float16* wvh = wkh + WN;
  _Float16* woh = wvh + WN;

  // one-time f32 -> f16 conversion + fragment-major swizzle
  const int cvtBlocks = (int)(WN / 1024);  // 4 elements/thread, 256 threads
  cvt_w_swizzle_kernel<<<cvtBlocks, 256, 0, stream>>>(wq, wqh);
  cvt_w_swizzle_kernel<<<cvtBlocks, 256, 0, stream>>>(wk, wkh);
  cvt_w_swizzle_kernel<<<cvtBlocks, 256, 0, stream>>>(wv, wvh);
  cvt_w_swizzle_kernel<<<cvtBlocks, 256, 0, stream>>>(wo, woh);

  const int smem1 = 32 * XLD * 2 + 32 * DIM_ * 4 + 32 * 8 * 4;  // ~290 KB
  const int smem3 = 32 * XLD * 2;                                // ~97 KB
  hipFuncSetAttribute((const void*)qkv_rmsnorm_rope_kernel,
                      hipFuncAttributeMaxDynamicSharedMemorySize, smem1);
  hipFuncSetAttribute((const void*)out_proj_kernel,
                      hipFuncAttributeMaxDynamicSharedMemorySize, smem3);

  qkv_rmsnorm_rope_kernel<<<(B_ * S_) / 32, 256, smem1, stream>>>(
      x, wqh, bq, wkh, bk, wvh, bv, nqw, nkw, cf, sf, ch, sh, cw, sw,
      qb, kb, vb);

  flash_attn_kernel<<<dim3(B_ * NH_, S_ / 64), 128, 0, stream>>>(
      qb, kb, vb, seq_lens, ab);

  out_proj_kernel<<<(B_ * S_) / 32, 256, smem3, stream>>>(
      ab, woh, bo, (float*)d_out);
}
